// vq_ffn_Block_79594333929820
// MI455X (gfx1250) — compile-verified, matrix operand order checked by reading, not run on hardware
//
#include <hip/hip_runtime.h>
#include <math.h>

// ---- problem dims ----
#define B_   32
#define N_   1024
#define D_   768
#define H_   3072
#define K_   1024
#define DC_  4
#define M_   (B_ * N_)          // 32768 tokens
#define BETA 0.25f
#define EPS_ 1e-5f

typedef __attribute__((ext_vector_type(16))) __bf16 v16bf;
typedef __attribute__((ext_vector_type(8)))  float  v8f;

// ---- workspace layout (bytes) ----
#define OFF_TLOSS 0
#define OFF_XNB   131072UL
#define OFF_W1T   (OFF_XNB + (size_t)M_ * D_ * 2)
#define OFF_W2T   (OFF_W1T + (size_t)H_ * D_ * 2)
#define OFF_HB    (OFF_W2T + (size_t)D_ * H_ * 2)

// 16x32 bf16 A/B fragment (row-major rows, contiguous K), per CDNA5 ISA:
// lanes 0-15: row = lane,    K = [0..7] + [16..23]
// lanes 16-31: row = lane-16, K = [8..15] + [24..31]
__device__ inline v16bf load_frag(const __bf16* __restrict__ p) {
  v16bf a;
  ((int4*)&a)[0] = *(const int4*)(p);        // global_load_b128
  ((int4*)&a)[1] = *(const int4*)(p + 16);
  return a;
}

__device__ inline v8f wmma_bf16(v16bf a, v16bf b, v8f c) {
  return __builtin_amdgcn_wmma_f32_16x16x32_bf16(false, a, false, b,
                                                 (short)0, c, false, false);
}

// ---------------- weight prep: fp32 -> bf16 transposed ----------------
__global__ __launch_bounds__(256) void prep_weights_kernel(
    const float* __restrict__ w1, const float* __restrict__ w2,
    unsigned short* __restrict__ w1t_raw, unsigned short* __restrict__ w2t_raw) {
  __bf16* w1t = reinterpret_cast<__bf16*>(w1t_raw);
  __bf16* w2t = reinterpret_cast<__bf16*>(w2t_raw);
  size_t total = (size_t)D_ * H_;
  size_t stride = (size_t)gridDim.x * blockDim.x;
  for (size_t i = (size_t)blockIdx.x * blockDim.x + threadIdx.x; i < total; i += stride) {
    int hh = (int)(i / D_);
    int dd = (int)(i % D_);
    w1t[(size_t)hh * D_ + dd] = (__bf16)w1[(size_t)dd * H_ + hh];  // w1t[h][d] = w1[d][h]
    w2t[(size_t)dd * H_ + hh] = (__bf16)w2[(size_t)hh * D_ + dd];  // w2t[d][h] = w2[h][d]
  }
}

// ---------------- VQ (down-proj, argmin, up-proj) + LayerNorm ----------------
// one wave32 per token; 8 tokens per 256-thread block
__global__ __launch_bounds__(256) void vq_ln_kernel(
    const float* __restrict__ x,    const float* __restrict__ win,
    const float* __restrict__ bin,  const float* __restrict__ codebook,
    const float* __restrict__ wout, const float* __restrict__ bout,
    const float* __restrict__ ln_g, const float* __restrict__ ln_b,
    unsigned short* __restrict__ xnb_raw, float* __restrict__ tloss) {
  __bf16* xnb = reinterpret_cast<__bf16*>(xnb_raw);
  const int lane = threadIdx.x & 31;
  const int wave = threadIdx.x >> 5;
  const int t = blockIdx.x * 8 + wave;
  const float* xr = x + (size_t)t * D_;

  // ze = x @ win + bin  (lane-strided over D, butterfly reduce)
  float a0 = 0.f, a1 = 0.f, a2 = 0.f, a3 = 0.f;
  for (int k = lane; k < D_; k += 32) {
    float xv = xr[k];
    float4 w = *(const float4*)(win + (size_t)k * DC_);
    a0 += xv * w.x; a1 += xv * w.y; a2 += xv * w.z; a3 += xv * w.w;
  }
#pragma unroll
  for (int off = 16; off >= 1; off >>= 1) {
    a0 += __shfl_xor(a0, off); a1 += __shfl_xor(a1, off);
    a2 += __shfl_xor(a2, off); a3 += __shfl_xor(a3, off);
  }
  const float ze0 = a0 + bin[0], ze1 = a1 + bin[1], ze2 = a2 + bin[2], ze3 = a3 + bin[3];

  // argmin over codebook (||ze - c||^2 has same argmin as reference distance)
  float best = 3.402823e38f;
  int bidx = 0x7fffffff;
  for (int j = lane; j < K_; j += 32) {
    float4 c = *(const float4*)(codebook + (size_t)j * DC_);
    float d0 = ze0 - c.x, d1 = ze1 - c.y, d2 = ze2 - c.z, d3 = ze3 - c.w;
    float dist = d0 * d0 + d1 * d1 + d2 * d2 + d3 * d3;
    if (dist < best || (dist == best && j < bidx)) { best = dist; bidx = j; }
  }
#pragma unroll
  for (int off = 16; off >= 1; off >>= 1) {
    float ob = __shfl_xor(best, off);
    int   oi = __shfl_xor(bidx, off);
    if (ob < best || (ob == best && oi < bidx)) { best = ob; bidx = oi; }
  }
  float4 cq = *(const float4*)(codebook + (size_t)bidx * DC_);
  if (lane == 0) {
    float d0 = ze0 - cq.x, d1 = ze1 - cq.y, d2 = ze2 - cq.z, d3 = ze3 - cq.w;
    tloss[t] = d0 * d0 + d1 * d1 + d2 * d2 + d3 * d3;
  }

  // xq = zq @ wout + bout (forward value of straight-through is zq), then LN
  float xv[24];
  float s = 0.f, ss = 0.f;
  int ii = 0;
  for (int d = lane; d < D_; d += 32, ++ii) {
    float v = bout[d] + cq.x * wout[d] + cq.y * wout[D_ + d]
                      + cq.z * wout[2 * D_ + d] + cq.w * wout[3 * D_ + d];
    xv[ii] = v; s += v; ss += v * v;
  }
#pragma unroll
  for (int off = 16; off >= 1; off >>= 1) {
    s += __shfl_xor(s, off); ss += __shfl_xor(ss, off);
  }
  const float mu = s * (1.0f / D_);
  const float var = ss * (1.0f / D_) - mu * mu;
  const float rstd = rsqrtf(var + EPS_);
  __bf16* orow = xnb + (size_t)t * D_;
  ii = 0;
  for (int d = lane; d < D_; d += 32, ++ii)
    orow[d] = (__bf16)((xv[ii] - mu) * rstd * ln_g[d] + ln_b[d]);
}

// ---------------- GEMM1: h = gelu(xn @ w1 + b1) ----------------
// one wave per 16(M) x 64(H) strip; double-buffered fragments so WMMAs
// overlap with the next K-step's loads (in-order LOADcnt => nonzero waits)
__global__ __launch_bounds__(256) void gemm1_gelu_kernel(
    const unsigned short* __restrict__ xnb_raw, const unsigned short* __restrict__ w1t_raw,
    const float* __restrict__ b1, unsigned short* __restrict__ hb_raw) {
  const __bf16* xnb = reinterpret_cast<const __bf16*>(xnb_raw);
  const __bf16* w1t = reinterpret_cast<const __bf16*>(w1t_raw);
  __bf16* hb = reinterpret_cast<__bf16*>(hb_raw);
  __shared__ __align__(16) __bf16 stage[8][16 * 64];

  const int lane = threadIdx.x & 31;
  const int wave = threadIdx.x >> 5;
  const int task = blockIdx.x * 8 + wave;
  const int m0 = (task / (H_ / 64)) * 16;
  const int n0 = (task % (H_ / 64)) * 64;

  const int r  = lane & 15;
  const int kb = (lane >> 4) * 8;
  const __bf16* ap = xnb + (size_t)(m0 + r) * D_ + kb;
  const __bf16* bp = w1t + (size_t)(n0 + r) * D_ + kb;

  v8f acc[4] = {v8f{}, v8f{}, v8f{}, v8f{}};
  v16bf aA, aB, bA[4], bB[4];

  // prologue: fragments for K-step 0
  aA = load_frag(ap);
#pragma unroll
  for (int j = 0; j < 4; ++j) bA[j] = load_frag(bp + (size_t)j * 16 * D_);

  for (int kk = 0; kk < D_ - 64; kk += 64) {
    aB = load_frag(ap + kk + 32);
#pragma unroll
    for (int j = 0; j < 4; ++j) bB[j] = load_frag(bp + (size_t)j * 16 * D_ + kk + 32);
#pragma unroll
    for (int j = 0; j < 4; ++j) acc[j] = wmma_bf16(aA, bA[j], acc[j]);

    aA = load_frag(ap + kk + 64);
#pragma unroll
    for (int j = 0; j < 4; ++j) bA[j] = load_frag(bp + (size_t)j * 16 * D_ + kk + 64);
#pragma unroll
    for (int j = 0; j < 4; ++j) acc[j] = wmma_bf16(aB, bB[j], acc[j]);
  }
  // epilogue K-steps: D_-64 and D_-32
  aB = load_frag(ap + D_ - 32);
#pragma unroll
  for (int j = 0; j < 4; ++j) bB[j] = load_frag(bp + (size_t)j * 16 * D_ + D_ - 32);
#pragma unroll
  for (int j = 0; j < 4; ++j) acc[j] = wmma_bf16(aA, bA[j], acc[j]);
#pragma unroll
  for (int j = 0; j < 4; ++j) acc[j] = wmma_bf16(aB, bB[j], acc[j]);

  // epilogue: +b1, exact gelu, stage to LDS, coalesced b128 stores
  const int col = lane & 15;
  const int rb  = (lane >> 4) * 8;
#pragma unroll
  for (int j = 0; j < 4; ++j) {
    const float bj = b1[n0 + j * 16 + col];
#pragma unroll
    for (int i = 0; i < 8; ++i) {
      float v = acc[j][i] + bj;
      float g = 0.5f * v * (1.0f + erff(v * 0.70710678118f));
      stage[wave][(rb + i) * 64 + j * 16 + col] = (__bf16)g;
    }
  }
  __syncthreads();
#pragma unroll
  for (int p = 0; p < 4; ++p) {
    const int row = p * 4 + (lane >> 3);
    const int c8  = (lane & 7) * 8;
    *(int4*)(hb + (size_t)(m0 + row) * H_ + n0 + c8) =
        *(const int4*)(&stage[wave][row * 64 + c8]);
  }
}

// ---------------- GEMM2: out = x + ls_gamma * (h @ w2 + b2) ----------------
__global__ __launch_bounds__(256) void gemm2_out_kernel(
    const unsigned short* __restrict__ hb_raw, const unsigned short* __restrict__ w2t_raw,
    const float* __restrict__ b2, const float* __restrict__ ls_gamma,
    const float* __restrict__ x, float* __restrict__ out) {
  const __bf16* hb  = reinterpret_cast<const __bf16*>(hb_raw);
  const __bf16* w2t = reinterpret_cast<const __bf16*>(w2t_raw);

  const int lane = threadIdx.x & 31;
  const int wave = threadIdx.x >> 5;
  const int task = blockIdx.x * 8 + wave;
  const int m0 = (task / (D_ / 64)) * 16;
  const int n0 = (task % (D_ / 64)) * 64;

  const int r  = lane & 15;
  const int kb = (lane >> 4) * 8;
  const __bf16* ap = hb  + (size_t)(m0 + r) * H_ + kb;
  const __bf16* bp = w2t + (size_t)(n0 + r) * H_ + kb;

  v8f acc[4] = {v8f{}, v8f{}, v8f{}, v8f{}};
  v16bf aA, aB, bA[4], bB[4];

  aA = load_frag(ap);
#pragma unroll
  for (int j = 0; j < 4; ++j) bA[j] = load_frag(bp + (size_t)j * 16 * H_);

  for (int kk = 0; kk < H_ - 64; kk += 64) {
    aB = load_frag(ap + kk + 32);
#pragma unroll
    for (int j = 0; j < 4; ++j) bB[j] = load_frag(bp + (size_t)j * 16 * H_ + kk + 32);
#pragma unroll
    for (int j = 0; j < 4; ++j) acc[j] = wmma_bf16(aA, bA[j], acc[j]);

    aA = load_frag(ap + kk + 64);
#pragma unroll
    for (int j = 0; j < 4; ++j) bA[j] = load_frag(bp + (size_t)j * 16 * H_ + kk + 64);
#pragma unroll
    for (int j = 0; j < 4; ++j) acc[j] = wmma_bf16(aB, bB[j], acc[j]);
  }
  aB = load_frag(ap + H_ - 32);
#pragma unroll
  for (int j = 0; j < 4; ++j) bB[j] = load_frag(bp + (size_t)j * 16 * H_ + H_ - 32);
#pragma unroll
  for (int j = 0; j < 4; ++j) acc[j] = wmma_bf16(aA, bA[j], acc[j]);
#pragma unroll
  for (int j = 0; j < 4; ++j) acc[j] = wmma_bf16(aB, bB[j], acc[j]);

  const int col = lane & 15;
  const int rb  = (lane >> 4) * 8;
#pragma unroll
  for (int j = 0; j < 4; ++j) {
    const int d = n0 + j * 16 + col;
    const float bb = b2[d];
    const float g  = ls_gamma[d];
#pragma unroll
    for (int i = 0; i < 8; ++i) {
      const int m = m0 + rb + i;
      out[(size_t)m * D_ + d] = x[(size_t)m * D_ + d] + g * (acc[j][i] + bb);
    }
  }
}

// ---------------- deterministic fixed-order loss reduction ----------------
__global__ __launch_bounds__(256) void reduce_loss_kernel(
    const float* __restrict__ tloss, float* __restrict__ out_loss) {
  __shared__ float sm[256];
  float s = 0.f;
  for (int i = threadIdx.x; i < M_; i += 256) s += tloss[i];
  sm[threadIdx.x] = s;
  __syncthreads();
  for (int off = 128; off >= 1; off >>= 1) {
    if ((int)threadIdx.x < off) sm[threadIdx.x] += sm[threadIdx.x + off];
    __syncthreads();
  }
  if (threadIdx.x == 0)
    *out_loss = sm[0] * ((1.0f + BETA) / (float)((size_t)M_ * DC_));
}

extern "C" void kernel_launch(void* const* d_in, const int* in_sizes, int n_in,
                              void* d_out, int out_size, void* d_ws, size_t ws_size,
                              hipStream_t stream) {
  (void)in_sizes; (void)n_in; (void)out_size; (void)ws_size;
  const float* x        = (const float*)d_in[0];
  const float* win      = (const float*)d_in[1];
  const float* bin      = (const float*)d_in[2];
  const float* codebook = (const float*)d_in[3];
  const float* wout     = (const float*)d_in[4];
  const float* bout     = (const float*)d_in[5];
  const float* ln_g     = (const float*)d_in[6];
  const float* ln_b     = (const float*)d_in[7];
  const float* w1       = (const float*)d_in[8];
  const float* b1       = (const float*)d_in[9];
  const float* w2       = (const float*)d_in[10];
  const float* b2       = (const float*)d_in[11];
  const float* ls_gamma = (const float*)d_in[12];

  float* out = (float*)d_out;                       // [M*D] out, then [1] loss
  char* ws = (char*)d_ws;
  float*          tloss = (float*)(ws + OFF_TLOSS);
  unsigned short* xnb   = (unsigned short*)(ws + OFF_XNB);
  unsigned short* w1t   = (unsigned short*)(ws + OFF_W1T);
  unsigned short* w2t   = (unsigned short*)(ws + OFF_W2T);
  unsigned short* hb    = (unsigned short*)(ws + OFF_HB);

  prep_weights_kernel<<<2048, 256, 0, stream>>>(w1, w2, w1t, w2t);
  vq_ln_kernel<<<M_ / 8, 256, 0, stream>>>(x, win, bin, codebook, wout, bout,
                                           ln_g, ln_b, xnb, tloss);
  gemm1_gelu_kernel<<<(M_ / 16) * (H_ / 64) / 8, 256, 0, stream>>>(xnb, w1t, b1, hb);
  gemm2_out_kernel<<<(M_ / 16) * (D_ / 64) / 8, 256, 0, stream>>>(hb, w2t, b2, ls_gamma,
                                                                  x, out);
  reduce_loss_kernel<<<1, 256, 0, stream>>>(tloss, out + (size_t)M_ * D_);
}
